// Model_64055142252668
// MI455X (gfx1250) — matrix operand order repaired and checked by reading, not compiled, not run on hardware
//
#include <hip/hip_runtime.h>

#define NSAMP 32768
#define TT    32768
#define MD    128
#define EV    128
#define NFR   128
#define WIN   512
#define PADT  33024
#define BB    4
#define PI2   6.2831853071795864769f
#define TILEH 5120   // f16 elements per packed weight tile: 128 rows x 40 (32 used + 8 pad)

typedef _Float16 f16;
typedef _Float16 v16h __attribute__((ext_vector_type(16)));
typedef float    v8f  __attribute__((ext_vector_type(8)));

// ---- WMMA fragment loaders (wave32, f16 16x16x32 layouts per CDNA5 ISA 7.12.2) ----
// Tiles stored [row][k] f16 with 16B-aligned rows so each fragment is 2x ds_load_b128.
union V16Q { v16h v; uint4 q[2]; };

// A tile: row stride = strideV uint4 (strideV*8 f16 per row).
__device__ __forceinline__ v16h frag_a4(const uint4* p0, int rowBase, int lane, int strideV) {
  int m = lane & 15, hf = lane >> 4;
  const uint4* p = p0 + (size_t)(rowBase + m) * strideV;
  V16Q r;
  r.q[0] = p[hf];          // k pairs {8*hf .. 8*hf+7}
  r.q[1] = p[2 + hf];      // k pairs {16+8*hf .. 16+8*hf+7}
  return r.v;
}
// B tile transposed [n][k]: row stride = strideV uint4.
__device__ __forceinline__ v16h frag_b4(const uint4* p0, int nBase, int lane, int strideV) {
  int n = nBase + (lane & 15), hf = lane >> 4;
  const uint4* p = p0 + (size_t)n * strideV;
  V16Q r;
  r.q[0] = p[hf];
  r.q[1] = p[2 + hf];
  return r.v;
}

// ---- CDNA5 async global->LDS block copy (ASYNCcnt), 10240B tile = 640 x b128 ----
__device__ __forceinline__ void async_copy_tile(const f16* __restrict__ g, unsigned ldsOff, int tid) {
#pragma unroll
  for (int it = 0; it < 3; ++it) {
    int c = it * 256 + tid;
    if (c < 640) {
      unsigned dst = ldsOff + c * 16;
      unsigned long long ga = (unsigned long long)(const void*)(g + c * 8);
      asm volatile("global_load_async_to_lds_b128 %0, %1, off" :: "v"(dst), "v"(ga) : "memory");
    }
  }
}
__device__ __forceinline__ void async_wait0() {
  asm volatile("s_wait_asynccnt 0x0" ::: "memory");
}

// ================= weight pre-pack kernels (f32 -> f16 padded LDS tile layout) =================
__global__ void k_pack_conv(const float* __restrict__ w, f16* __restrict__ out) {
  // w: (7,128,128,3) -> out: (7,3,4,128,40)
  long i = (long)blockIdx.x * 256 + threadIdx.x;
  if (i >= 7L * 3 * 4 * 128 * 32) return;
  int kk = (int)(i & 31);
  int o  = (int)((i >> 5) & 127);
  int ks = (int)((i >> 12) & 3);
  long r = i >> 14;
  int tap = (int)(r % 3);
  int l   = (int)(r / 3);
  out[(((size_t)(l * 3 + tap) * 4 + ks) * 128 + o) * 40 + kk] =
      (f16)w[(((size_t)l * 128 + o) * 128 + ks * 32 + kk) * 3 + tap];
}
__global__ void k_pack_mat(const float* __restrict__ w, f16* __restrict__ out) {
  // w: (7,128,128) -> out: (7,4,128,40)
  long i = (long)blockIdx.x * 256 + threadIdx.x;
  if (i >= 7L * 4 * 128 * 32) return;
  int kk = (int)(i & 31);
  int o  = (int)((i >> 5) & 127);
  int ks = (int)((i >> 12) & 3);
  int l  = (int)(i >> 14);
  out[(((size_t)l * 4 + ks) * 128 + o) * 40 + kk] = (f16)w[((size_t)l * 128 + o) * 128 + ks * 32 + kk];
}
__global__ void k_pack_fb(const float* __restrict__ w, f16* __restrict__ out) {
  // w: (128,512) -> out: (16,128,40)
  int i = blockIdx.x * 256 + threadIdx.x; // 16*128*32 = 65536
  int kk = i & 31; int c = (i >> 5) & 127; int ks = i >> 12;
  out[((size_t)ks * 128 + c) * 40 + kk] = (f16)w[c * 512 + ks * 32 + kk];
}

// ================= Front-end conv: n[b,c,t] = sum_k x[b, t+k-256]*fbw[c,k] =================
__global__ __launch_bounds__(256) void k_fbconv(const float* __restrict__ x,
                                                const f16* __restrict__ fbpk,
                                                float* __restrict__ nout) {
  __shared__ uint4 sm4[1280];            // A: 640 (128x20dw), B: 640
  uint4* A4 = sm4;
  uint4* Bq = sm4 + 640;
  f16* Af = (f16*)A4;
  unsigned offB = (unsigned)(size_t)(void*)Bq;
  int wg = blockIdx.x; int b = wg >> 8; int t0 = (wg & 255) << 7;
  int tid = threadIdx.x; int wave = tid >> 5; int lane = tid & 31;
  v8f acc[8];
  v8f z8 = {0.f,0.f,0.f,0.f,0.f,0.f,0.f,0.f};
#pragma unroll
  for (int j = 0; j < 8; ++j) acc[j] = z8;
  const float* xb = x + (size_t)b * TT;
  for (int ks = 0; ks < 16; ++ks) {
    async_copy_tile(fbpk + (size_t)ks * TILEH, offB, tid);
    for (int idx = tid; idx < 4096; idx += 256) {
      int kk = idx >> 7; int tl = idx & 127;
      int xi = t0 + tl + ks * 32 + kk - 256;
      float v = (xi >= 0 && xi < TT) ? xb[xi] : 0.0f;
      Af[tl * 40 + kk] = (f16)v;
    }
    async_wait0();
    __syncthreads();
    v16h a = frag_a4(A4, wave * 16, lane, 5);
#pragma unroll
    for (int nt = 0; nt < 8; ++nt) {
      v16h bb = frag_b4(Bq, nt * 16, lane, 5);
      acc[nt] = __builtin_amdgcn_wmma_f32_16x16x32_f16(false, a, false, bb, (short)0, acc[nt], false, false);
    }
    __syncthreads();
  }
  int hf = lane >> 4, ncl = lane & 15;
#pragma unroll
  for (int nt = 0; nt < 8; ++nt)
#pragma unroll
    for (int r = 0; r < 8; ++r) {
      int tl = wave * 16 + 8 * hf + r; int c = nt * 16 + ncl;
      nout[((size_t)b * MD + c) * TT + t0 + tl] = acc[nt][r];
    }
}

// ================= Fused dilated WaveNet layer =================
__global__ __launch_bounds__(256) void k_dilated(const float* __restrict__ nin, float* __restrict__ nout,
                                                 float* __restrict__ accg,
                                                 const f16* __restrict__ spk, const float* __restrict__ sb,
                                                 const f16* __restrict__ gpk, const float* __restrict__ gb,
                                                 const f16* __restrict__ opk, const float* __restrict__ ob,
                                                 const f16* __restrict__ npk, const float* __restrict__ nb,
                                                 int d, int initAcc) {
  // LDS plan (uint4 units): H: [0,2176) (128 x 68dw) ; A overlays [0,640) (dead before H written)
  //                          B0: [2176,2816) ; B1: [2816,3456)   -> 55296 B total
  __shared__ uint4 sm4[3456];
  uint4* A4  = sm4;
  uint4* H4  = sm4;
  uint4* B0q = sm4 + 2176;
  uint4* B1q = sm4 + 2816;
  f16* Af = (f16*)A4;
  f16* Hf = (f16*)H4;
  unsigned offB0 = (unsigned)(size_t)(void*)B0q;
  unsigned offB1 = (unsigned)(size_t)(void*)B1q;
  int wg = blockIdx.x; int b = wg >> 8; int t0 = (wg & 255) << 7;
  int tid = threadIdx.x; int wave = tid >> 5; int lane = tid & 31;
  int hf = lane >> 4, ncl = lane & 15;
  v8f z8 = {0.f,0.f,0.f,0.f,0.f,0.f,0.f,0.f};
  v8f sac[8], gac[8];
#pragma unroll
  for (int j = 0; j < 8; ++j) { sac[j] = z8; gac[j] = z8; }
  // ---- stage 1: s,g = 3-tap dilated convs (6 GEMM-taps, K=128) ----
  for (int tap = 0; tap < 3; ++tap) {
    int xoff = (tap - 1) * d;
    for (int ks = 0; ks < 4; ++ks) {
      int k0 = ks * 32;
      async_copy_tile(spk + (size_t)(tap * 4 + ks) * TILEH, offB0, tid);
      async_copy_tile(gpk + (size_t)(tap * 4 + ks) * TILEH, offB1, tid);
      for (int idx = tid; idx < 4096; idx += 256) {
        int kk = idx >> 7; int tl = idx & 127;
        int tt = t0 + tl + xoff;
        float v = (tt >= 0 && tt < TT) ? nin[((size_t)b * MD + k0 + kk) * TT + tt] : 0.0f;
        Af[tl * 40 + kk] = (f16)v;
      }
      async_wait0();
      __syncthreads();
      v16h a = frag_a4(A4, wave * 16, lane, 5);
#pragma unroll
      for (int nt = 0; nt < 8; ++nt) {
        v16h bs = frag_b4(B0q, nt * 16, lane, 5);
        v16h bg = frag_b4(B1q, nt * 16, lane, 5);
        sac[nt] = __builtin_amdgcn_wmma_f32_16x16x32_f16(false, a, false, bs, (short)0, sac[nt], false, false);
        gac[nt] = __builtin_amdgcn_wmma_f32_16x16x32_f16(false, a, false, bg, (short)0, gac[nt], false, false);
      }
      __syncthreads();
    }
  }
  // ---- gated activation -> LDS h tile (f16, stride 136) ----
#pragma unroll
  for (int nt = 0; nt < 8; ++nt)
#pragma unroll
    for (int r = 0; r < 8; ++r) {
      int c = nt * 16 + ncl; int tl = wave * 16 + 8 * hf + r;
      float s = sac[nt][r] + sb[c];
      float g = gac[nt][r] + gb[c];
      float hv = tanhf(s) * (1.0f / (1.0f + __expf(-g)));
      Hf[tl * 136 + c] = (f16)hv;
    }
  __syncthreads();
  // ---- stage 2: acc += h@out_w.T ; n_next = h@next_w.T + skip ----
  v8f oac[8], nac[8];
#pragma unroll
  for (int j = 0; j < 8; ++j) { oac[j] = z8; nac[j] = z8; }
  for (int ks = 0; ks < 4; ++ks) {
    async_copy_tile(opk + (size_t)ks * TILEH, offB0, tid);
    async_copy_tile(npk + (size_t)ks * TILEH, offB1, tid);
    async_wait0();
    __syncthreads();
    v16h a = frag_a4(H4 + 4 * ks, wave * 16, lane, 17);
#pragma unroll
    for (int nt = 0; nt < 8; ++nt) {
      v16h bo = frag_b4(B0q, nt * 16, lane, 5);
      v16h bn = frag_b4(B1q, nt * 16, lane, 5);
      oac[nt] = __builtin_amdgcn_wmma_f32_16x16x32_f16(false, a, false, bo, (short)0, oac[nt], false, false);
      nac[nt] = __builtin_amdgcn_wmma_f32_16x16x32_f16(false, a, false, bn, (short)0, nac[nt], false, false);
    }
    __syncthreads();
  }
#pragma unroll
  for (int nt = 0; nt < 8; ++nt)
#pragma unroll
    for (int r = 0; r < 8; ++r) {
      int c = nt * 16 + ncl; int tl = wave * 16 + 8 * hf + r;
      size_t gi = ((size_t)b * MD + c) * TT + t0 + tl;
      float av = oac[nt][r] + ob[c];
      accg[gi] = initAcc ? av : (accg[gi] + av);
      nout[gi] = nac[nt][r] + nb[c] + nin[gi];
    }
}

// ================= norms / agg-max / top-k / latent gather =================
__global__ void k_norms(const float* __restrict__ xf, float* __restrict__ norms) {
  int i = blockIdx.x * 256 + threadIdx.x; // b*T + t
  int b = i >> 15; int t = i & 32767;
  float s = 0.f;
  for (int c = 0; c < MD; ++c) { float v = xf[((size_t)b * MD + c) * TT + t]; s += v * v; }
  norms[i] = sqrtf(s);
}

__global__ void k_aggmax(const float* __restrict__ xf, float* __restrict__ agg) {
  int bc = blockIdx.x; int tid = threadIdx.x;
  __shared__ float sm[256];
  float m = -1e30f;
  for (int t = tid; t < TT; t += 256) m = fmaxf(m, xf[(size_t)bc * TT + t]);
  sm[tid] = m; __syncthreads();
  for (int s = 128; s > 0; s >>= 1) { if (tid < s) sm[tid] = fmaxf(sm[tid], sm[tid + s]); __syncthreads(); }
  if (tid == 0) agg[bc] = sm[0];
}

__global__ void k_topk(float* __restrict__ norms, float* __restrict__ vals, int* __restrict__ idxs) {
  int b = blockIdx.x; int tid = threadIdx.x;
  __shared__ float sv[256]; __shared__ int si[256];
  float* nb = norms + (size_t)b * TT;
  for (int a = 0; a < 32; ++a) {
    float bv = -1e30f; int bi = TT;
    for (int t = tid; t < TT; t += 256) {
      float v = nb[t];
      if (v > bv || (v == bv && t < bi)) { bv = v; bi = t; }
    }
    sv[tid] = bv; si[tid] = bi; __syncthreads();
    for (int s = 128; s > 0; s >>= 1) {
      if (tid < s) {
        if (sv[tid + s] > sv[tid] || (sv[tid + s] == sv[tid] && si[tid + s] < si[tid])) {
          sv[tid] = sv[tid + s]; si[tid] = si[tid + s];
        }
      }
      __syncthreads();
    }
    if (tid == 0) { vals[b * 32 + a] = sv[0]; idxs[b * 32 + a] = si[0]; nb[si[0]] = -1e30f; }
    __syncthreads();
  }
}

__global__ void k_latents(const float* __restrict__ xf, const int* __restrict__ idxs, float* __restrict__ lat) {
  int i = blockIdx.x * 256 + threadIdx.x; // e*128 + c
  int e = i >> 7; int c = i & 127; int b = e >> 5; int a = e & 31;
  lat[i] = xf[((size_t)b * MD + c) * TT + idxs[b * 32 + a]];
}

// ================= generic dense layer: out[r,o] = act(in[r,:]·w[o,:] + b[o]) =================
__global__ void k_linleaky(const float* __restrict__ in, const float* __restrict__ w, const float* __restrict__ bi,
                           float* __restrict__ out, int rows, int K, int Nout, int act) {
  int i = blockIdx.x * 256 + threadIdx.x;
  if (i >= rows * Nout) return;
  int r = i / Nout, o = i - r * Nout;
  float acc = bi[o];
  const float* ip = in + (size_t)r * K;
  const float* wp = w + (size_t)o * K;
  for (int k = 0; k < K; ++k) acc += ip[k] * wp[k];
  out[i] = (act && acc < 0.f) ? 0.2f * acc : acc;
}

// ================= repeat(2) + conv1d(128->128, k=3, pad=1) + leaky =================
__global__ void k_upconv(const float* __restrict__ in, const float* __restrict__ w, const float* __restrict__ bi,
                         float* __restrict__ out, int L) {
  int twoL = 2 * L;
  long i = (long)blockIdx.x * 256 + threadIdx.x;
  long tot = (long)EV * MD * twoL;
  if (i >= tot) return;
  int t = (int)(i % twoL); long rem = i / twoL;
  int o = (int)(rem % MD); int e = (int)(rem / MD);
  float acc = bi[o];
  for (int ic = 0; ic < MD; ++ic) {
    const float* wp = w + (size_t)(o * MD + ic) * 3;
    const float* ip = in + ((size_t)e * MD + ic) * L;
    for (int kp = 0; kp < 3; ++kp) {
      int u = t + kp - 1;
      if (u >= 0 && u < twoL) acc += ip[u >> 1] * wp[kp];
    }
  }
  out[i] = (acc < 0.f) ? 0.2f * acc : acc;
}

__global__ void k_finenv(const float* __restrict__ in, const float* __restrict__ w, const float* __restrict__ bi,
                         float* __restrict__ envs) {
  int i = blockIdx.x * 256 + threadIdx.x; // e*128 + t
  int e = i >> 7; int t = i & 127;
  float acc = bi[0];
  for (int ic = 0; ic < MD; ++ic) {
    const float* ip = in + ((size_t)e * MD + ic) * NFR;
    const float* wp = w + ic * 3;
    for (int kp = 0; kp < 3; ++kp) { int u = t + kp - 1; if (u >= 0 && u < NFR) acc += ip[u] * wp[kp]; }
  }
  envs[i] = acc * acc;
}

__global__ void k_fintf(const float* __restrict__ in, const float* __restrict__ w, const float* __restrict__ bi,
                        float* __restrict__ trans) {
  long i = (long)blockIdx.x * 256 + threadIdx.x; // e*514*128 + j*128 + t
  long tot = (long)EV * 514 * NFR;
  if (i >= tot) return;
  int t = (int)(i % NFR); long rem = i / NFR;
  int j = (int)(rem % 514); int e = (int)(rem / 514);
  float acc = bi[j];
  for (int ic = 0; ic < MD; ++ic) {
    const float* ip = in + ((size_t)e * MD + ic) * NFR;
    const float* wp = w + (size_t)(j * MD + ic) * 3;
    for (int kp = 0; kp < 3; ++kp) { int u = t + kp - 1; if (u >= 0 && u < NFR) acc += ip[u] * wp[kp]; }
  }
  trans[i] = acc;
}

// ================= env linear upsample 128 -> 32768 (+zero pad to 33024) =================
__global__ void k_upsample(const float* __restrict__ envs, float* __restrict__ envup) {
  long i = (long)blockIdx.x * 256 + threadIdx.x;
  if (i >= (long)EV * PADT) return;
  int t = (int)(i % PADT); int e = (int)(i / PADT);
  float v = 0.f;
  if (t < NSAMP) {
    float src = (t + 0.5f) * (128.0f / 32768.0f) - 0.5f;
    src = fminf(fmaxf(src, 0.0f), 127.0f);
    int i0 = (int)floorf(src); int i1 = min(i0 + 1, 127);
    float fr = src - (float)i0;
    v = envs[e * NFR + i0] * (1.0f - fr) + envs[e * NFR + i1] * fr;
  }
  envup[i] = v;
}

// ================= DFT helpers (ortho norms cancel; use std rfft/irfft pair) =================
__global__ void k_noisespec(const float* __restrict__ noise, float* __restrict__ nspec) {
  int k = threadIdx.x;
  if (k > 256) return;
  float dc, ds; __sincosf(PI2 * (float)k / 512.0f, &ds, &dc);
  float c = 1.0f, s = 0.0f, ar = 0.0f, ai = 0.0f;
  for (int t = 0; t < WIN; ++t) {
    float v = noise[t];
    ar += v * c; ai -= v * s;
    float cn = c * dc - s * ds; s = s * dc + c * ds; c = cn;
  }
  nspec[2 * k] = ar; nspec[2 * k + 1] = ai;
}

__global__ void k_impulse(const float* __restrict__ nspec, const float* __restrict__ imptf, float* __restrict__ impl) {
  int i = blockIdx.x * 256 + threadIdx.x; // e*512 + t
  int e = i >> 9; int t = i & 511;
  const float* itf = imptf + (size_t)e * 514;
  float x0 = nspec[0] * itf[0] - nspec[1] * itf[1];
  float xN = nspec[512] * itf[512] - nspec[513] * itf[513];
  float acc = x0 + ((t & 1) ? -xN : xN);
  float dc, ds; __sincosf(PI2 * (float)t / 512.0f, &ds, &dc);
  float c = dc, s = ds;
  for (int k = 1; k < 256; ++k) {
    float nr = nspec[2 * k], ni = nspec[2 * k + 1];
    float tr = itf[2 * k], ti = itf[2 * k + 1];
    float xr = nr * tr - ni * ti;
    float xi = nr * ti + ni * tr;
    acc += 2.0f * (xr * c - xi * s);
    float cn = c * dc - s * ds; s = s * dc + c * ds; c = cn;
  }
  impl[i] = acc * (1.0f / 512.0f);
}

__global__ void k_htime(const float* __restrict__ trans, float* __restrict__ ht) {
  long i = (long)blockIdx.x * 256 + threadIdx.x; // e*128*512 + f*512 + t
  int t = (int)(i & 511); long ef = i >> 9; int f = (int)(ef & 127); int e = (int)(ef >> 7);
  const float* Tp = trans + (size_t)e * 514 * NFR + f;
  float x0 = Tp[0];
  float xN = Tp[(size_t)512 * NFR];
  float acc = x0 + ((t & 1) ? -xN : xN);
  float dc, ds; __sincosf(PI2 * (float)t / 512.0f, &ds, &dc);
  float c = dc, s = ds;
  for (int k = 1; k < 256; ++k) {
    float xr = Tp[(size_t)(2 * k) * NFR];
    float xi = Tp[(size_t)(2 * k + 1) * NFR];
    acc += 2.0f * (xr * c - xi * s);
    float cn = c * dc - s * ds; s = s * dc + c * ds; c = cn;
  }
  ht[i] = acc * (1.0f / 512.0f);
}

// ================= sequential per-event frame scan (circular conv in LDS) + overlap-add =================
__global__ __launch_bounds__(512) void k_scan(const float* __restrict__ impl, const float* __restrict__ envup,
                                              const float* __restrict__ ht, float* __restrict__ atomsb) {
  int e = blockIdx.x; int t = threadIdx.x;
  __shared__ float nxt[WIN], hds[WIN], carry[WIN];
  carry[t] = 0.0f;
  float imp = impl[(size_t)e * WIN + t];
  float ham = 0.54f - 0.46f * __cosf(PI2 * (float)t / 512.0f);
  __syncthreads();
  for (int f = 0; f < NFR; ++f) {
    nxt[t] = imp * envup[(size_t)e * PADT + f * 256 + t] + carry[t];
    hds[t] = ht[((size_t)e * NFR + f) * WIN + t];
    __syncthreads();
    float a = 0.0f;
    for (int tau = 0; tau < WIN; ++tau) a += nxt[tau] * hds[(t - tau) & 511];
    a *= ham;
    __syncthreads();
    carry[t] = a;
    atomsb[(size_t)e * PADT + f * 256 + t] += a;
    __syncthreads();
  }
}

// ================= scatter-as-gather: final[b,t] = sum_a atoms[b,a,t-idx]*val =================
__global__ void k_final(const float* __restrict__ atomsb, const float* __restrict__ topv,
                        const int* __restrict__ topi, float* __restrict__ finalb) {
  int i = blockIdx.x * 256 + threadIdx.x; // b*N + t
  int b = i >> 15; int t = i & 32767;
  float s = 0.f;
  for (int a = 0; a < 32; ++a) {
    int idx = topi[b * 32 + a];
    if (t >= idx) s += atomsb[(size_t)(b * 32 + a) * PADT + (t - idx)] * topv[b * 32 + a];
  }
  finalb[i] = s;
}

__global__ void k_softmax8(const float* __restrict__ r, float* __restrict__ sr) {
  int b = threadIdx.x;
  if (b >= BB) return;
  float mx = -1e30f;
  for (int q = 0; q < 8; ++q) mx = fmaxf(mx, r[b * 8 + q]);
  float s = 0.f, ev[8];
  for (int q = 0; q < 8; ++q) { ev[q] = __expf(r[b * 8 + q] - mx); s += ev[q]; }
  for (int q = 0; q < 8; ++q) sr[b * 8 + q] = ev[q] / s;
}

__global__ void k_ir(const float* __restrict__ sr, const float* __restrict__ rooms, float* __restrict__ ir) {
  int i = blockIdx.x * 256 + threadIdx.x; // b*N + t
  int b = i >> 15; int t = i & 32767;
  float s = 0.f;
  for (int q = 0; q < 8; ++q) s += sr[b * 8 + q] * rooms[(size_t)q * NSAMP + t];
  ir[i] = s;
}

// ================= reverb: wet[t] = sum_{k<=t} final[k]*ir[t-k], tiled through LDS =================
__global__ void k_wet(const float* __restrict__ finalb, const float* __restrict__ ir, float* __restrict__ wet) {
  int blk = blockIdx.x; int b = blk >> 7; int t0 = (blk & 127) << 8;
  int tid = threadIdx.x;
  __shared__ float sf[256]; __shared__ float sir[512];
  float acc = 0.f;
  for (int k0 = 0; k0 <= t0; k0 += 256) {
    sf[tid] = finalb[(size_t)b * NSAMP + k0 + tid];
    int base = t0 - k0 - 255;
    int g0 = base + tid;        sir[tid]       = (g0 >= 0 && g0 < NSAMP) ? ir[(size_t)b * NSAMP + g0] : 0.f;
    int g1 = base + tid + 256;  sir[tid + 256] = (g1 >= 0 && g1 < NSAMP) ? ir[(size_t)b * NSAMP + g1] : 0.f;
    __syncthreads();
    int jmax = (k0 == t0) ? tid : 255;
    for (int j = 0; j <= jmax; ++j) acc += sf[j] * sir[tid + 255 - j];
    __syncthreads();
  }
  wet[(size_t)b * NSAMP + t0 + tid] = acc;
}

__global__ void k_out(const float* __restrict__ wet, const float* __restrict__ finalb,
                      const float* __restrict__ mval, float* __restrict__ out) {
  int i = blockIdx.x * 256 + threadIdx.x;
  int b = i >> 15;
  float m = mval[b];
  out[i] = m * wet[i] + finalb[i] * (1.0f - m);
}

// ================================== host ==================================
extern "C" void kernel_launch(void* const* d_in, const int* in_sizes, int n_in,
                              void* d_out, int out_size, void* d_ws, size_t ws_size,
                              hipStream_t stream) {
  (void)in_sizes; (void)n_in; (void)out_size; (void)ws_size;
  const float* x    = (const float*)d_in[0];
  const float* fbw  = (const float*)d_in[1];
  const float* dsw  = (const float*)d_in[2];
  const float* dsb  = (const float*)d_in[3];
  const float* dgw  = (const float*)d_in[4];
  const float* dgb  = (const float*)d_in[5];
  const float* dow  = (const float*)d_in[6];
  const float* dob  = (const float*)d_in[7];
  const float* dnw  = (const float*)d_in[8];
  const float* dnb  = (const float*)d_in[9];
  const float* elw  = (const float*)d_in[10];
  const float* elb  = (const float*)d_in[11];
  const float* ecw  = (const float*)d_in[12];
  const float* ecb  = (const float*)d_in[13];
  const float* efw  = (const float*)d_in[14];
  const float* efb  = (const float*)d_in[15];
  const float* tlw  = (const float*)d_in[16];
  const float* tlb  = (const float*)d_in[17];
  const float* tcw  = (const float*)d_in[18];
  const float* tcb  = (const float*)d_in[19];
  const float* tfw  = (const float*)d_in[20];
  const float* tfb  = (const float*)d_in[21];
  const float* iw   = (const float*)d_in[22];
  const float* ibw  = (const float*)d_in[23];
  const float* iow  = (const float*)d_in[24];
  const float* iob  = (const float*)d_in[25];
  const float* rw   = (const float*)d_in[26];
  const float* rbv  = (const float*)d_in[27];
  const float* rowt = (const float*)d_in[28];
  const float* rob  = (const float*)d_in[29];
  const float* mw   = (const float*)d_in[30];
  const float* mbv  = (const float*)d_in[31];
  const float* mow  = (const float*)d_in[32];
  const float* mob  = (const float*)d_in[33];
  const float* rooms = (const float*)d_in[34];
  const float* noise = (const float*)d_in[35];

  size_t off = 0;
  auto alloc = [&](size_t n) -> float* {
    float* p = (float*)d_ws + off;
    off += (n + 63) & ~(size_t)63;
    return p;
  };
  const size_t BCT = (size_t)BB * MD * TT;
  float* n0     = alloc(BCT);
  float* n1     = alloc(BCT);
  float* accb   = alloc(BCT);
  float* norms  = alloc((size_t)BB * NSAMP);
  float* topv   = alloc(BB * 32);
  int*   topi   = (int*)alloc(BB * 32);
  float* lat    = alloc(EV * MD);
  float* ub0    = alloc((size_t)EV * MD * NFR);
  float* ub1    = alloc((size_t)EV * MD * NFR);
  float* trans  = alloc((size_t)EV * 514 * NFR);
  float* envs   = alloc(EV * NFR);
  float* envup  = alloc((size_t)EV * PADT);
  float* imptf  = alloc(EV * 514);
  float* impl   = alloc(EV * WIN);
  float* htime  = alloc((size_t)EV * NFR * WIN);
  float* atomsb = alloc((size_t)EV * PADT);
  float* finalb = alloc((size_t)BB * NSAMP);
  float* aggb   = alloc(BB * MD);
  float* tmpa   = alloc(BB * MD);
  float* tmpb   = alloc(BB * MD);
  float* r8     = alloc(BB * 8);
  float* sr8    = alloc(BB * 8);
  float* mval   = alloc(BB);
  float* irb    = alloc((size_t)BB * NSAMP);
  float* wetb   = alloc((size_t)BB * NSAMP);
  float* nspec  = alloc(2 * 257);
  float* lt0    = alloc(EV * MD);
  float* lt1    = alloc(EV * MD);
  // packed f16 weight tiles (async-copy source, LDS tile layout incl. 40-f16 row stride)
  f16* fbpk = (f16*)alloc(40960);   // 16 tiles
  f16* spk  = (f16*)alloc(215040);  // 7*3*4 tiles
  f16* gpk  = (f16*)alloc(215040);
  f16* opk  = (f16*)alloc(71680);   // 7*4 tiles
  f16* npk  = (f16*)alloc(71680);

  // pre-pack weights to f16 tiles
  k_pack_fb<<<256, 256, 0, stream>>>(fbw, fbpk);
  k_pack_conv<<<1344, 256, 0, stream>>>(dsw, spk);
  k_pack_conv<<<1344, 256, 0, stream>>>(dgw, gpk);
  k_pack_mat<<<448, 256, 0, stream>>>(dow, opk);
  k_pack_mat<<<448, 256, 0, stream>>>(dnw, npk);

  // front-end conv (WMMA GEMM, K=512)
  k_fbconv<<<1024, 256, 0, stream>>>(x, fbpk, n0);

  // dilated stack (WMMA, fused, async weight tiles)
  static const int dil[7] = {1, 3, 9, 27, 81, 243, 1};
  float* cur = n0; float* nx = n1;
  for (int i = 0; i < 7; ++i) {
    k_dilated<<<1024, 256, 0, stream>>>(cur, nx, accb,
        spk + (size_t)i * 12 * TILEH, dsb + i * MD,
        gpk + (size_t)i * 12 * TILEH, dgb + i * MD,
        opk + (size_t)i * 4 * TILEH, dob + i * MD,
        npk + (size_t)i * 4 * TILEH, dnb + i * MD,
        dil[i], i == 0 ? 1 : 0);
    float* tsw = cur; cur = nx; nx = tsw;
  }

  k_norms<<<512, 256, 0, stream>>>(accb, norms);
  k_aggmax<<<512, 256, 0, stream>>>(accb, aggb);
  k_topk<<<4, 256, 0, stream>>>(norms, topv, topi);
  k_latents<<<64, 256, 0, stream>>>(accb, topi, lat);

  // env branch
  k_linleaky<<<512, 256, 0, stream>>>(lat, elw, elb, ub0, EV, MD, 1024, 0);
  {
    float* cin = ub0; float* cout = ub1; int L = 8;
    for (int li = 0; li < 4; ++li) {
      long tot = (long)EV * MD * 2 * L;
      k_upconv<<<(int)((tot + 255) / 256), 256, 0, stream>>>(cin, ecw + (size_t)li * MD * MD * 3, ecb + li * MD, cout, L);
      float* tsw = cin; cin = cout; cout = tsw; L *= 2;
    }
  }
  k_finenv<<<64, 256, 0, stream>>>(ub0, efw, efb, envs);
  k_upsample<<<(int)(((long)EV * PADT + 255) / 256), 256, 0, stream>>>(envs, envup);

  // transfer branch
  k_linleaky<<<512, 256, 0, stream>>>(lat, tlw, tlb, ub0, EV, MD, 1024, 0);
  {
    float* cin = ub0; float* cout = ub1; int L = 8;
    for (int li = 0; li < 4; ++li) {
      long tot = (long)EV * MD * 2 * L;
      k_upconv<<<(int)((tot + 255) / 256), 256, 0, stream>>>(cin, tcw + (size_t)li * MD * MD * 3, tcb + li * MD, cout, L);
      float* tsw = cin; cin = cout; cout = tsw; L *= 2;
    }
  }
  k_fintf<<<(int)(((long)EV * 514 * NFR + 255) / 256), 256, 0, stream>>>(ub0, tfw, tfb, trans);

  // impulse transfer stack
  k_linleaky<<<64, 256, 0, stream>>>(lat, iw, ibw, lt0, EV, MD, MD, 1);
  k_linleaky<<<64, 256, 0, stream>>>(lt0, iw + MD * MD, ibw + MD, lt1, EV, MD, MD, 1);
  k_linleaky<<<64, 256, 0, stream>>>(lt1, iw + 2 * MD * MD, ibw + 2 * MD, lt0, EV, MD, MD, 1);
  k_linleaky<<<(EV * 514 + 255) / 256, 256, 0, stream>>>(lt0, iow, iob, imptf, EV, MD, 514, 0);

  // event synthesis
  k_noisespec<<<1, 288, 0, stream>>>(noise, nspec);
  k_impulse<<<256, 256, 0, stream>>>(nspec, imptf, impl);
  k_htime<<<32768, 256, 0, stream>>>(trans, htime);
  hipMemsetAsync(atomsb, 0, (size_t)EV * PADT * sizeof(float), stream);
  k_scan<<<128, 512, 0, stream>>>(impl, envup, htime, atomsb);
  k_final<<<512, 256, 0, stream>>>(atomsb, topv, topi, finalb);

  // room / mix heads
  k_linleaky<<<2, 256, 0, stream>>>(aggb, rw, rbv, tmpa, BB, MD, MD, 1);
  k_linleaky<<<2, 256, 0, stream>>>(tmpa, rw + MD * MD, rbv + MD, tmpb, BB, MD, MD, 1);
  k_linleaky<<<2, 256, 0, stream>>>(tmpb, rw + 2 * MD * MD, rbv + 2 * MD, tmpa, BB, MD, MD, 1);
  k_linleaky<<<1, 256, 0, stream>>>(tmpa, rowt, rob, r8, BB, MD, 8, 0);
  k_linleaky<<<2, 256, 0, stream>>>(aggb, mw, mbv, tmpa, BB, MD, MD, 1);
  k_linleaky<<<2, 256, 0, stream>>>(tmpa, mw + MD * MD, mbv + MD, tmpb, BB, MD, MD, 1);
  k_linleaky<<<2, 256, 0, stream>>>(tmpb, mw + 2 * MD * MD, mbv + 2 * MD, tmpa, BB, MD, MD, 1);
  k_linleaky<<<1, 256, 0, stream>>>(tmpa, mow, mob, mval, BB, MD, 1, 0);

  k_softmax8<<<1, 32, 0, stream>>>(r8, sr8);
  k_ir<<<512, 256, 0, stream>>>(sr8, rooms, irb);
  k_wet<<<512, 256, 0, stream>>>(finalb, irb, wetb);
  k_out<<<512, 256, 0, stream>>>(wetb, finalb, mval, (float*)d_out);
}
